// GCN_Dense_7627861917705
// MI455X (gfx1250) — matrix operand order repaired and checked by reading, not compiled, not run on hardware
//
#include <hip/hip_runtime.h>
#include <hip/hip_bf16.h>

// ---------------------------------------------------------------------------
// Types for CDNA5 WMMA (wave32): v_wmma_f32_16x16x32_bf16
// ---------------------------------------------------------------------------
typedef __bf16 v16bf __attribute__((ext_vector_type(16)));
typedef float  v8f   __attribute__((ext_vector_type(8)));

union Frag { uint4 u[2]; v16bf v; };   // 32 bytes = 8 VGPRs per lane

#define LDT 40   // padded LDS row stride (bf16 elements): 80B, 16B-aligned

// ---------------------------------------------------------------------------
// fp32 -> bf16 (round to nearest even) + pack helpers
// ---------------------------------------------------------------------------
__device__ inline unsigned short f2bf(float f) {
    unsigned int u = __float_as_uint(f);
    unsigned int lsb = (u >> 16) & 1u;
    u += 0x7FFFu + lsb;
    return (unsigned short)(u >> 16);
}
__device__ inline unsigned int pack2bf(float a, float b) {
    return (unsigned int)f2bf(a) | ((unsigned int)f2bf(b) << 16);
}

// ---------------------------------------------------------------------------
// CSR build step 1: integer degrees
// ---------------------------------------------------------------------------
__global__ __launch_bounds__(256) void degree_kernel(
    const int* __restrict__ erow, int* __restrict__ deg, int e) {
    int i = blockIdx.x * 256 + threadIdx.x;
    if (i < e) atomicAdd(&deg[erow[i]], 1);
}

// ---------------------------------------------------------------------------
// CSR build step 2: exclusive prefix sum over n rows (single block, 1024 thr)
// rowPtr[0..n], rowPtr[n] = total
// ---------------------------------------------------------------------------
__global__ __launch_bounds__(1024) void scan_kernel(
    const int* __restrict__ deg, int* __restrict__ rowPtr, int n) {
    __shared__ int sums[1024];
    const int t = threadIdx.x;
    const int chunk = (n + 1023) >> 10;
    int beg = t * chunk;
    int end = min(beg + chunk, n);
    int s = 0;
    for (int i = beg; i < end && i < n; ++i) s += deg[i];
    sums[t] = s;
    __syncthreads();
    for (int off = 1; off < 1024; off <<= 1) {
        int v = (t >= off) ? sums[t - off] : 0;
        __syncthreads();
        sums[t] += v;
        __syncthreads();
    }
    int base = (t == 0) ? 0 : sums[t - 1];
    for (int i = beg; i < end && i < n; ++i) {
        rowPtr[i] = base;
        base += deg[i];
    }
    if (t == 1023) rowPtr[n] = sums[1023];
}

// ---------------------------------------------------------------------------
// CSR build step 3: cursor init + bucket fill of column indices
// ---------------------------------------------------------------------------
__global__ __launch_bounds__(256) void copy_int_kernel(
    const int* __restrict__ src, int* __restrict__ dst, int n) {
    int i = blockIdx.x * 256 + threadIdx.x;
    if (i < n) dst[i] = src[i];
}
__global__ __launch_bounds__(256) void fill_csr_kernel(
    const int* __restrict__ erow, const int* __restrict__ ecol,
    int* __restrict__ cursor, int* __restrict__ colIdx, int e) {
    int i = blockIdx.x * 256 + threadIdx.x;
    if (i < e) {
        int p = atomicAdd(&cursor[erow[i]], 1);
        colIdx[p] = ecol[i];
    }
}

// ---------------------------------------------------------------------------
// X (n x 300, f32) -> Xb (n x 320, bf16, zero-padded K)
// ---------------------------------------------------------------------------
__global__ __launch_bounds__(256) void conv_x_kernel(
    const float* __restrict__ x, unsigned short* __restrict__ xb,
    int n, int cin, int kp) {
    long long idx = (long long)blockIdx.x * 256 + threadIdx.x;
    if (idx >= (long long)n * kp) return;
    int i = (int)(idx / kp), k = (int)(idx % kp);
    xb[idx] = (k < cin) ? f2bf(x[(long long)i * cin + k]) : (unsigned short)0;
}

// ---------------------------------------------------------------------------
// W1 (300 x 2048, f32) -> W1b (320 x 2048, bf16, zero-padded rows)
// ---------------------------------------------------------------------------
__global__ __launch_bounds__(256) void conv_w1_kernel(
    const float* __restrict__ w, unsigned short* __restrict__ wb,
    int cin, int kp, int nCol) {
    long long idx = (long long)blockIdx.x * 256 + threadIdx.x;
    if (idx >= (long long)kp * nCol) return;
    int k = (int)(idx / nCol), c = (int)(idx % nCol);
    wb[idx] = (k < cin) ? f2bf(w[(long long)k * nCol + c]) : (unsigned short)0;
}

// ---------------------------------------------------------------------------
// generic f32 -> bf16 copy (W2)
// ---------------------------------------------------------------------------
__global__ __launch_bounds__(256) void conv_kernel(
    const float* __restrict__ src, unsigned short* __restrict__ dst, long long total) {
    long long idx = (long long)blockIdx.x * 256 + threadIdx.x;
    if (idx < total) dst[idx] = f2bf(src[idx]);
}

// ---------------------------------------------------------------------------
// Tiled bf16 WMMA GEMM: C(MxN, f32) = A(Mxlda bf16) * B(KxN bf16)
// Block tile 128x128, 256 threads = 8 waves; wave tile 32x64 = 2x4 WMMA tiles.
// K multiple of 32 (padded inputs); N multiple of 128.
// ---------------------------------------------------------------------------
__global__ __launch_bounds__(256) void gemm_bf16_wmma(
    const unsigned short* __restrict__ A,
    const unsigned short* __restrict__ B,
    float* __restrict__ C,
    int M, int N, int K, int lda) {
    __shared__ unsigned short lsA[128 * LDT];
    __shared__ unsigned short lsB[128 * LDT];

    const int tid   = threadIdx.x;
    const int lane  = tid & 31;
    const int wave  = tid >> 5;
    const int waveM = wave & 3;     // 32-row slab within 128
    const int waveN = wave >> 2;    // 64-col slab within 128
    const int halfA = lane >> 4;    // 0/1 (lane group)
    const int l16   = lane & 15;

    const int blockM = blockIdx.y * 128;
    const int blockN = blockIdx.x * 128;

    const int aRow  = tid >> 1;     // 0..127
    const int aHalf = tid & 1;      // which 16-elem half of the 32-wide K slice
    const int bK    = tid >> 3;     // 0..31
    const int bCG   = tid & 7;      // 16-col group

    v8f acc[2][4];
    #pragma unroll
    for (int i = 0; i < 2; ++i)
        #pragma unroll
        for (int j = 0; j < 4; ++j)
            #pragma unroll
            for (int v = 0; v < 8; ++v) acc[i][j][v] = 0.0f;

    for (int k0 = 0; k0 < K; k0 += 32) {
        // ---- stage A tile (128 x 32) into LDS, row-major ----
        uint4 a0 = {0, 0, 0, 0}, a1 = {0, 0, 0, 0};
        {
            int gr = blockM + aRow;
            if (gr < M) {
                const uint4* src =
                    (const uint4*)(A + (size_t)gr * lda + k0 + aHalf * 16);
                a0 = src[0];
                a1 = src[1];
                if (k0 + 32 < K) __builtin_prefetch(src + 8, 0, 0);  // global_prefetch_b8
            }
        }
        *(uint4*)&lsA[aRow * LDT + aHalf * 16]     = a0;
        *(uint4*)&lsA[aRow * LDT + aHalf * 16 + 8] = a1;

        // ---- stage B tile (32 x 128) transposed into LDS: lsB[col*LDT + k] ----
        {
            const unsigned short* bsrc =
                B + (size_t)(k0 + bK) * N + blockN + bCG * 16;
            uint4 b0 = *(const uint4*)bsrc;
            uint4 b1 = *(const uint4*)(bsrc + 8);
            unsigned short tmp[16];
            *(uint4*)&tmp[0] = b0;
            *(uint4*)&tmp[8] = b1;
            #pragma unroll
            for (int j = 0; j < 16; ++j)
                lsB[(bCG * 16 + j) * LDT + bK] = tmp[j];
        }
        __syncthreads();

        // ---- load fragments (ISA 16-bit A 16x32 / B 32x16 VGPR layouts) ----
        Frag af[2], bfr[4];
        #pragma unroll
        for (int tm = 0; tm < 2; ++tm) {
            int r = waveM * 32 + tm * 16 + l16;
            int base = r * LDT + halfA * 8;          // K = halfA*8 .. +8
            af[tm].u[0] = *(const uint4*)&lsA[base];
            af[tm].u[1] = *(const uint4*)&lsA[base + 16];  // K = 16+halfA*8 ..
        }
        #pragma unroll
        for (int tn = 0; tn < 4; ++tn) {
            int c = waveN * 64 + tn * 16 + l16;
            int base = c * LDT + halfA * 8;          // K = halfA*8 .. +8, 16+halfA*8 ..
            bfr[tn].u[0] = *(const uint4*)&lsB[base];
            bfr[tn].u[1] = *(const uint4*)&lsB[base + 16];
        }

        // ---- 8x v_wmma_f32_16x16x32_bf16 ----
        #pragma unroll
        for (int tm = 0; tm < 2; ++tm)
            #pragma unroll
            for (int tn = 0; tn < 4; ++tn)
                acc[tm][tn] = __builtin_amdgcn_wmma_f32_16x16x32_bf16(
                    false, af[tm].v, false, bfr[tn].v,
                    (short)0, acc[tm][tn], false, false);

        __syncthreads();
    }

    // ---- store: C/D layout: VGPR v -> row = base + halfA*8 + v, col = l16 ----
    #pragma unroll
    for (int tm = 0; tm < 2; ++tm) {
        int r0 = blockM + waveM * 32 + tm * 16 + halfA * 8;
        #pragma unroll
        for (int tn = 0; tn < 4; ++tn) {
            int c = blockN + waveN * 64 + tn * 16 + l16;
            #pragma unroll
            for (int v = 0; v < 8; ++v) {
                int r = r0 + v;
                if (r < M) C[(size_t)r * N + c] = acc[tm][tn][v];
            }
        }
    }
}

// ---------------------------------------------------------------------------
// SPMM (CSR, one block per destination row) fused with +b1, LeakyReLU, ->bf16
// out_bf16[row] = bf16( leakyrelu( (1/max(deg,1)) * sum_j src[col_j] + b1 ) )
// 256 threads x 2 float4 = full 2048-wide row in registers; no atomics.
// ---------------------------------------------------------------------------
__global__ __launch_bounds__(256) void spmm_act_kernel(
    const int* __restrict__ rowPtr, const int* __restrict__ colIdx,
    const float* __restrict__ src, const float* __restrict__ bias,
    unsigned short* __restrict__ out, int C) {
    const int row = blockIdx.x;
    const int beg = rowPtr[row], end = rowPtr[row + 1];
    const int t = threadIdx.x;

    float4 a0 = {0, 0, 0, 0}, a1 = {0, 0, 0, 0};
    for (int e = beg; e < end; ++e) {
        const float4* s = (const float4*)(src + (size_t)colIdx[e] * C);
        float4 v0 = s[t];
        float4 v1 = s[t + 256];
        a0.x += v0.x; a0.y += v0.y; a0.z += v0.z; a0.w += v0.w;
        a1.x += v1.x; a1.y += v1.y; a1.z += v1.z; a1.w += v1.w;
    }
    const float w = 1.0f / fmaxf((float)(end - beg), 1.0f);
    const float4 b0 = ((const float4*)bias)[t];
    const float4 b1v = ((const float4*)bias)[t + 256];

    float v[8];
    v[0] = a0.x * w + b0.x;  v[1] = a0.y * w + b0.y;
    v[2] = a0.z * w + b0.z;  v[3] = a0.w * w + b0.w;
    v[4] = a1.x * w + b1v.x; v[5] = a1.y * w + b1v.y;
    v[6] = a1.z * w + b1v.z; v[7] = a1.w * w + b1v.w;
    #pragma unroll
    for (int i = 0; i < 8; ++i) v[i] = (v[i] >= 0.0f) ? v[i] : 0.2f * v[i];

    uint2* o = (uint2*)(out + (size_t)row * C);
    uint2 o0, o1;
    o0.x = pack2bf(v[0], v[1]); o0.y = pack2bf(v[2], v[3]);
    o1.x = pack2bf(v[4], v[5]); o1.y = pack2bf(v[6], v[7]);
    o[t]       = o0;
    o[t + 256] = o1;
}

// ---------------------------------------------------------------------------
// SPMM (CSR) fused with +b2 and row-wise L2 normalization -> final output
// ---------------------------------------------------------------------------
__global__ __launch_bounds__(256) void spmm_norm_kernel(
    const int* __restrict__ rowPtr, const int* __restrict__ colIdx,
    const float* __restrict__ src, const float* __restrict__ bias,
    float* __restrict__ out, int C) {
    __shared__ float red[256];
    const int row = blockIdx.x;
    const int beg = rowPtr[row], end = rowPtr[row + 1];
    const int t = threadIdx.x;

    float4 a0 = {0, 0, 0, 0}, a1 = {0, 0, 0, 0};
    for (int e = beg; e < end; ++e) {
        const float4* s = (const float4*)(src + (size_t)colIdx[e] * C);
        float4 v0 = s[t];
        float4 v1 = s[t + 256];
        a0.x += v0.x; a0.y += v0.y; a0.z += v0.z; a0.w += v0.w;
        a1.x += v1.x; a1.y += v1.y; a1.z += v1.z; a1.w += v1.w;
    }
    const float w = 1.0f / fmaxf((float)(end - beg), 1.0f);
    const float4 b0 = ((const float4*)bias)[t];
    const float4 b1v = ((const float4*)bias)[t + 256];

    a0.x = a0.x * w + b0.x;  a0.y = a0.y * w + b0.y;
    a0.z = a0.z * w + b0.z;  a0.w = a0.w * w + b0.w;
    a1.x = a1.x * w + b1v.x; a1.y = a1.y * w + b1v.y;
    a1.z = a1.z * w + b1v.z; a1.w = a1.w * w + b1v.w;

    float ss = a0.x * a0.x + a0.y * a0.y + a0.z * a0.z + a0.w * a0.w
             + a1.x * a1.x + a1.y * a1.y + a1.z * a1.z + a1.w * a1.w;
    red[t] = ss;
    __syncthreads();
    for (int s = 128; s > 0; s >>= 1) {
        if (t < s) red[t] += red[t + s];
        __syncthreads();
    }
    const float inv = 1.0f / fmaxf(sqrtf(red[0]), 1e-12f);

    float4* o = (float4*)(out + (size_t)row * C);
    float4 r0, r1;
    r0.x = a0.x * inv; r0.y = a0.y * inv; r0.z = a0.z * inv; r0.w = a0.w * inv;
    r1.x = a1.x * inv; r1.y = a1.y * inv; r1.z = a1.z * inv; r1.w = a1.w * inv;
    o[t]       = r0;
    o[t + 256] = r1;
}

// ---------------------------------------------------------------------------
// host launcher
// ---------------------------------------------------------------------------
extern "C" void kernel_launch(void* const* d_in, const int* in_sizes, int n_in,
                              void* d_out, int out_size, void* d_ws, size_t ws_size,
                              hipStream_t stream) {
    const float* x   = (const float*)d_in[0];
    const float* W1  = (const float*)d_in[1];
    const float* b1  = (const float*)d_in[2];
    const float* W2  = (const float*)d_in[3];
    const float* b2  = (const float*)d_in[4];
    const int* erow  = (const int*)d_in[5];
    const int* ecol  = (const int*)d_in[6];

    const int CIN = 300, KP = 320, CH = 2048, CO = 2048;
    const int N = in_sizes[0] / CIN;   // 50000
    const int E = in_sizes[5];         // 1600000

    // workspace carve-out (256B aligned slices)
    char* ws = (char*)d_ws;
    size_t off = 0;
    auto carve = [&](size_t bytes) -> void* {
        void* p = ws + off;
        off = (off + bytes + 255) & ~(size_t)255;
        return p;
    };
    int*            deg    = (int*)carve((size_t)N * 4);
    int*            rowPtr = (int*)carve((size_t)(N + 1) * 4);
    int*            cursor = (int*)carve((size_t)N * 4);
    int*            colIdx = (int*)carve((size_t)E * 4);
    unsigned short* Xb     = (unsigned short*)carve((size_t)N * KP * 2);
    unsigned short* W1b    = (unsigned short*)carve((size_t)KP * CH * 2);
    unsigned short* W2b    = (unsigned short*)carve((size_t)CH * CO * 2);
    unsigned short* Hb     = (unsigned short*)carve((size_t)N * CH * 2);
    float*          bufA   = (float*)carve((size_t)N * CH * 4);
    (void)ws_size; (void)n_in; (void)out_size;

    // 1) CSR build: degrees -> prefix sum -> bucket fill
    hipMemsetAsync(deg, 0, (size_t)N * 4, stream);
    degree_kernel<<<(E + 255) / 256, 256, 0, stream>>>(erow, deg, E);
    scan_kernel<<<1, 1024, 0, stream>>>(deg, rowPtr, N);
    copy_int_kernel<<<(N + 255) / 256, 256, 0, stream>>>(rowPtr, cursor, N);
    fill_csr_kernel<<<(E + 255) / 256, 256, 0, stream>>>(erow, ecol, cursor, colIdx, E);

    // 2) bf16 conversions (with K padding for GEMM1)
    {
        long long t = (long long)N * KP;
        conv_x_kernel<<<(unsigned)((t + 255) / 256), 256, 0, stream>>>(x, Xb, N, CIN, KP);
    }
    {
        long long t = (long long)KP * CH;
        conv_w1_kernel<<<(unsigned)((t + 255) / 256), 256, 0, stream>>>(W1, W1b, CIN, KP, CH);
    }
    {
        long long t = (long long)CH * CO;
        conv_kernel<<<(unsigned)((t + 255) / 256), 256, 0, stream>>>(W2, W2b, t);
    }

    // 3) GEMM1: bufA = Xb (N x 320) * W1b (320 x 2048)   [WMMA bf16]
    {
        dim3 grid(CH / 128, (N + 127) / 128);
        gemm_bf16_wmma<<<grid, 256, 0, stream>>>(Xb, W1b, bufA, N, CH, KP, KP);
    }

    // 4) SPMM1 + b1 + LeakyReLU -> Hb (bf16), no atomics
    spmm_act_kernel<<<N, 256, 0, stream>>>(rowPtr, colIdx, bufA, b1, Hb, CH);

    // 5) GEMM2: bufA = Hb (N x 2048) * W2b (2048 x 2048)   [WMMA bf16]
    {
        dim3 grid(CO / 128, (N + 127) / 128);
        gemm_bf16_wmma<<<grid, 256, 0, stream>>>(Hb, W2b, bufA, N, CO, CH, CH);
    }

    // 6) SPMM2 + b2 + row L2 normalize -> d_out
    spmm_norm_kernel<<<N, 256, 0, stream>>>(rowPtr, colIdx, bufA, b2, (float*)d_out, CO);
}
